// SimilarityLoss_60473139528480
// MI455X (gfx1250) — matrix operand order repaired and checked by reading, not compiled, not run on hardware
//
#include <hip/hip_runtime.h>
#include <hip/hip_bf16.h>

typedef __attribute__((ext_vector_type(16))) _Float16 v16h;
typedef __attribute__((ext_vector_type(8)))  _Float16 v8h;
typedef __attribute__((ext_vector_type(4)))  _Float16 v4h;
typedef __attribute__((ext_vector_type(8)))  float    v8f;
typedef __attribute__((ext_vector_type(4)))  float    v4f;

#define NB    4096
#define TWO_N 8192
#define DIM   512
#define LAM   0.5f
#define LOG2E 1.4426950408889634f
#define LN2   0.6931471805599453f

// d_ws layout: [0..255] float accumulators (acc[0]=negatives sum, acc[1]=positives sum)
//              [256 ...] reps f16 matrix, TWO_N x DIM = 8 MB

// softplus in log2 domain, raw hardware trans ops (no libm range fixups):
//   log2(1 + 2^(x*log2e))  ;  caller multiplies the SUM by ln2 once.
// Valid here: x = sim - lam in [-1.5, 0.5] -> log argument in [1.22, 2.65],
// far from denormal/cancellation regimes, so bare v_exp_f32/v_log_f32 suffice.
__device__ __forceinline__ float softplus_log2(float x) {
    float t = __builtin_amdgcn_exp2f(x * LOG2E);
    return __builtin_amdgcn_logf(1.0f + t);
}

// ---------------------------------------------------------------------------
// Kernel 1: L2-normalize rows of emb_i / emb_j, emit f16 reps. Also zeroes acc.
// ---------------------------------------------------------------------------
__global__ void __launch_bounds__(128)
simloss_normalize_kernel(const float* __restrict__ emb_i,
                         const float* __restrict__ emb_j,
                         _Float16* __restrict__ reps,
                         float* __restrict__ acc) {
    const int row = blockIdx.x;
    const int tid = threadIdx.x;
    if (row == 0 && tid < 8) acc[tid] = 0.0f;

    const float* src = (row < NB) ? (emb_i + (size_t)row * DIM)
                                  : (emb_j + (size_t)(row - NB) * DIM);
    v4f x = *(const v4f*)(src + tid * 4);
    float ss = x.x * x.x + x.y * x.y + x.z * x.z + x.w * x.w;
    #pragma unroll
    for (int off = 16; off > 0; off >>= 1) ss += __shfl_down(ss, off, 32);

    __shared__ float red[4];
    __shared__ float inv;
    const int lane = tid & 31, wid = tid >> 5;
    if (lane == 0) red[wid] = ss;
    __syncthreads();
    if (tid == 0) inv = rsqrtf(red[0] + red[1] + red[2] + red[3]);
    __syncthreads();

    const float r = inv;
    v4h o;
    o.x = (_Float16)(x.x * r);
    o.y = (_Float16)(x.y * r);
    o.z = (_Float16)(x.z * r);
    o.w = (_Float16)(x.w * r);
    *(v4h*)(reps + (size_t)row * DIM + tid * 4) = o;
}

// ---------------------------------------------------------------------------
// Kernel 2: positives. p_k = dot(reps[k], reps[k+NB]); acc[1] += softplus(-p+lam)
// (sim_ij == sim_ji, so accumulate once per k and divide by NB at the end.)
// ---------------------------------------------------------------------------
__global__ void __launch_bounds__(128)
simloss_positives_kernel(const _Float16* __restrict__ reps,
                         float* __restrict__ acc) {
    const int k   = blockIdx.x;
    const int tid = threadIdx.x;
    v4h av = *(const v4h*)(reps + (size_t)k * DIM + tid * 4);
    v4h bv = *(const v4h*)(reps + (size_t)(k + NB) * DIM + tid * 4);
    float s = (float)av.x * (float)bv.x + (float)av.y * (float)bv.y +
              (float)av.z * (float)bv.z + (float)av.w * (float)bv.w;
    #pragma unroll
    for (int off = 16; off > 0; off >>= 1) s += __shfl_down(s, off, 32);

    __shared__ float red[4];
    const int lane = tid & 31, wid = tid >> 5;
    if (lane == 0) red[wid] = s;
    __syncthreads();
    if (tid == 0) {
        float p = red[0] + red[1] + red[2] + red[3];
        atomicAdd(&acc[1], LN2 * softplus_log2(-p + LAM));
    }
}

// ---------------------------------------------------------------------------
// Kernel 3: fused sim = R*R^T via v_wmma_f32_16x16x32_f16; epilogue applies
// diag mask + softplus(s - lam) (log2 domain) and block-reduces into acc[0].
// sim is exactly symmetric, so only blocks with bx >= by run; off-diagonal
// blocks contribute x2. Block = 256 threads (8 waves); wave tile = 32x64
// (2x4 WMMA accumulators); block tile = 128x128; grid = 64x64 upper triangle.
// ---------------------------------------------------------------------------
__global__ void __launch_bounds__(256)
simloss_gemm_kernel(const _Float16* __restrict__ reps,
                    float* __restrict__ acc) {
    if (blockIdx.x < blockIdx.y) return;   // symmetry: skip lower triangle
    const float weight = (blockIdx.x == blockIdx.y) ? 1.0f : 2.0f;

    const int lane    = threadIdx.x & 31;
    const int wave    = threadIdx.x >> 5;   // 0..7
    const int waveRow = wave >> 1;          // 0..3
    const int waveCol = wave & 1;           // 0..1
    const int rowBase = blockIdx.y * 128 + waveRow * 32;
    const int colBase = blockIdx.x * 128 + waveCol * 64;

    // A 16x32 layout: lanes 0-15 -> M=lane, K=0..15;
    //                 lanes 16-31 -> M=lane-16, K=16..31
    const int mA    = lane & 15;
    const int kaOff = (lane >> 4) << 4;     // 0 or 16
    // B 32x16 layout: lanes 0-15 -> N=lane, K=0..15; lanes 16-31 -> N=lane-16, K=16..31
    const int nB    = lane & 15;
    const int kbOff = (lane >> 4) << 4;     // 0 or 16

    const _Float16* arow0 = reps + (size_t)(rowBase + mA) * DIM + kaOff;
    const _Float16* arow1 = arow0 + (size_t)16 * DIM;
    const _Float16* brow[4];
    #pragma unroll
    for (int ct = 0; ct < 4; ++ct)
        brow[ct] = reps + (size_t)(colBase + ct * 16 + nB) * DIM + kbOff;

    v8f c[2][4] = {};

    for (int k0 = 0; k0 < DIM; k0 += 32) {
        if (k0 + 32 < DIM) {
            __builtin_prefetch(arow0 + k0 + 32, 0, 3);
            __builtin_prefetch(brow[0] + k0 + 32, 0, 3);
        }
        v16h a[2];
        #pragma unroll
        for (int rt = 0; rt < 2; ++rt) {
            const _Float16* p = (rt ? arow1 : arow0) + k0;
            v8h lo = *(const v8h*)p;          // K = k0+kaOff .. +7
            v8h hi = *(const v8h*)(p + 8);    // K = k0+kaOff+8 .. +15
            #pragma unroll
            for (int i = 0; i < 8; ++i) { a[rt][i] = lo[i]; a[rt][i + 8] = hi[i]; }
        }
        v16h b[4];
        #pragma unroll
        for (int ct = 0; ct < 4; ++ct)
            b[ct] = *(const v16h*)(brow[ct] + k0);  // 32 contiguous bytes

        #pragma unroll
        for (int rt = 0; rt < 2; ++rt)
            #pragma unroll
            for (int ct = 0; ct < 4; ++ct)
                c[rt][ct] = __builtin_amdgcn_wmma_f32_16x16x32_f16(
                    false, a[rt], false, b[ct], (short)0, c[rt][ct], false, false);
    }

    // Epilogue: C/D layout: VGPR r holds M=r (lanes 0-15) / M=r+8 (lanes 16-31),
    // N = lane&15. Per element: fma, v_exp_f32, add, v_log_f32, cndmask, add.
    // Accumulate in log2 domain; fold ln2 (and the symmetry weight) into one
    // per-thread multiply before the reduction.
    float s = 0.0f;
    const int rOff = (lane >> 4) << 3;  // 0 or 8
    const int nCol = lane & 15;
    const float kmul = LOG2E;           // x*log2e
    const float kadd = -LAM * LOG2E;    // -lam*log2e folded into the fma
    #pragma unroll
    for (int rt = 0; rt < 2; ++rt)
        #pragma unroll
        for (int ct = 0; ct < 4; ++ct)
            #pragma unroll
            for (int r = 0; r < 8; ++r) {
                const int p = rowBase + rt * 16 + rOff + r;
                const int q = colBase + ct * 16 + nCol;
                float t = __builtin_amdgcn_exp2f(fmaf(c[rt][ct][r], kmul, kadd));
                float f = __builtin_amdgcn_logf(1.0f + t);
                s += (p == q) ? 0.0f : f;
            }
    s *= weight * LN2;

    #pragma unroll
    for (int off = 16; off > 0; off >>= 1) s += __shfl_down(s, off, 32);
    __shared__ float red[8];
    if (lane == 0) red[wave] = s;
    __syncthreads();
    if (threadIdx.x == 0) {
        float t = 0.0f;
        #pragma unroll
        for (int i = 0; i < 8; ++i) t += red[i];
        atomicAdd(&acc[0], t);
    }
}

// ---------------------------------------------------------------------------
// Kernel 4: finalize. out = pos_sum/N + neg_sum/(12N)
// ---------------------------------------------------------------------------
__global__ void simloss_finalize_kernel(const float* __restrict__ acc,
                                        float* __restrict__ out) {
    out[0] = acc[1] / (float)NB + acc[0] / (float)(12 * NB);
}

extern "C" void kernel_launch(void* const* d_in, const int* in_sizes, int n_in,
                              void* d_out, int out_size, void* d_ws, size_t ws_size,
                              hipStream_t stream) {
    const float* emb_i = (const float*)d_in[0];
    const float* emb_j = (const float*)d_in[1];
    float* out = (float*)d_out;
    float* acc = (float*)d_ws;
    _Float16* reps = (_Float16*)((char*)d_ws + 256);

    simloss_normalize_kernel<<<TWO_N, 128, 0, stream>>>(emb_i, emb_j, reps, acc);
    simloss_positives_kernel<<<NB, 128, 0, stream>>>(reps, acc);
    dim3 grid(TWO_N / 128, TWO_N / 128);
    simloss_gemm_kernel<<<grid, 256, 0, stream>>>(reps, acc);
    simloss_finalize_kernel<<<1, 1, 0, stream>>>(acc, out);
}
